// Block_61186104099473
// MI455X (gfx1250) — compile-verified, matrix operand order checked by reading, not run on hardware
//
#include <hip/hip_runtime.h>
#include <hip/hip_bf16.h>

typedef __attribute__((ext_vector_type(16))) _Float16 v16h;
typedef __attribute__((ext_vector_type(8)))  _Float16 v8h;
typedef __attribute__((ext_vector_type(8)))  float    v8f;
typedef __attribute__((ext_vector_type(4)))  int      v4i;

#define D_MODEL 768
#define NHEAD   12
#define DHEAD   64
#define TSEQ    2048

// ---- gfx1250 async global->LDS copy path (probe-verified this toolchain) ---
#if defined(__has_builtin)
#  if __has_builtin(__builtin_amdgcn_global_load_async_to_lds_b128)
#    define HAVE_ASYNC_LDS 1
#  endif
#  if __has_builtin(__builtin_amdgcn_s_wait_asynccnt)
#    define HAVE_WAIT_ASYNC 1
#  endif
#endif
#ifndef HAVE_ASYNC_LDS
#define HAVE_ASYNC_LDS 0
#endif
#ifndef HAVE_WAIT_ASYNC
#define HAVE_WAIT_ASYNC 0
#endif
#define ASG __attribute__((address_space(1)))
#define ASL __attribute__((address_space(3)))

static __device__ __forceinline__ v8f wmma_f16(v16h a, v16h b, v8f c) {
  return __builtin_amdgcn_wmma_f32_16x16x32_f16(false, a, false, b, (short)0, c,
                                                false, false);
}

// Two contiguous 16B runs -> one WMMA fragment (forces b128 codegen)
static __device__ __forceinline__ v16h load2x8(const _Float16* p0,
                                               const _Float16* p1) {
  v8h lo = *(const v8h*)p0;
  v8h hi = *(const v8h*)p1;
  return __builtin_shufflevector(lo, hi, 0, 1, 2, 3, 4, 5, 6, 7,
                                 8, 9, 10, 11, 12, 13, 14, 15);
}

static __device__ __forceinline__ float rowmax16(float v) {
  #pragma unroll
  for (int m = 1; m < 16; m <<= 1) v = fmaxf(v, __shfl_xor(v, m, 32));
  return v;
}
static __device__ __forceinline__ float rowsum16(float v) {
  #pragma unroll
  for (int m = 1; m < 16; m <<= 1) v += __shfl_xor(v, m, 32);
  return v;
}

// ---------------------------------------------------------------------------
// Transposing weight conversion: W (K,N) f32 -> Wt (N,K) f16
// ---------------------------------------------------------------------------
__global__ void transpose_w_kernel(const float* __restrict__ in,
                                   _Float16* __restrict__ out, int K, int N) {
  int i = blockIdx.x * blockDim.x + threadIdx.x;
  if (i >= K * N) return;
  int k = i / N, n = i % N;
  out[(size_t)n * K + k] = (_Float16)in[i];
}

// ---------------------------------------------------------------------------
// LayerNorm over D=768, one 256-thread block per row; emits f32 + f16
// ---------------------------------------------------------------------------
__global__ void layernorm_kernel(const float* __restrict__ x,
                                 const float* __restrict__ g,
                                 const float* __restrict__ b,
                                 float* __restrict__ y,
                                 _Float16* __restrict__ y16) {
  __shared__ float red[256];
  const int t = blockIdx.x;
  const float* xr = x + (size_t)t * D_MODEL;
  float s = 0.f, sq = 0.f;
  for (int i = threadIdx.x; i < D_MODEL; i += 256) {
    float v = xr[i];
    s += v; sq += v * v;
  }
  red[threadIdx.x] = s; __syncthreads();
  for (int st = 128; st > 0; st >>= 1) {
    if (threadIdx.x < st) red[threadIdx.x] += red[threadIdx.x + st];
    __syncthreads();
  }
  float mean = red[0] / D_MODEL;
  __syncthreads();
  red[threadIdx.x] = sq; __syncthreads();
  for (int st = 128; st > 0; st >>= 1) {
    if (threadIdx.x < st) red[threadIdx.x] += red[threadIdx.x + st];
    __syncthreads();
  }
  float var = red[0] / D_MODEL - mean * mean;
  float rstd = rsqrtf(var + 1e-5f);
  for (int i = threadIdx.x; i < D_MODEL; i += 256) {
    float v = (xr[i] - mean) * rstd * g[i] + b[i];
    y[(size_t)t * D_MODEL + i] = v;
    y16[(size_t)t * D_MODEL + i] = (_Float16)v;
  }
}

// ---------------------------------------------------------------------------
// WMMA GEMM, double-buffered LDS staging via async global->LDS copies.
//   C(MxN,f32) = A(MxK,f16 row-major) @ Bt(NxK,f16)^T + bias [+gelu][+res]
// Workgroup = 4 waves, 64x64 tile; each wave 16 rows x 64 cols (4 accs).
// Tile k+1 is copied by the async engine while tile k runs on WMMA.
// ---------------------------------------------------------------------------
#define GBM 64
#define GBN 64
__global__ void gemm_kernel(const _Float16* __restrict__ A,
                            const _Float16* __restrict__ Bt,
                            const float* __restrict__ bias,
                            const float* __restrict__ residual,
                            float* __restrict__ C,
                            _Float16* __restrict__ C16,
                            int M, int N, int K, int act) {
  __shared__ _Float16 As[2][GBM * 32];
  __shared__ _Float16 Bs[2][GBN * 32];
  const int tid = threadIdx.x;        // 128 threads
  const int wave = tid >> 5;
  const int lane = tid & 31;
  const int nloc = lane & 15;
  const int hi8 = (lane >> 4) << 3;
  const size_t bm = (size_t)blockIdx.x * GBM;
  const size_t bn = (size_t)blockIdx.y * GBN;
  v8f acc[4] = {};

  // Each thread owns two fixed 16B chunks of the A-slab and two of the B-slab.
  const int srow = tid >> 2;               // 0..31
  const int sseg = (tid & 3) << 3;         // half-offset 0/8/16/24
  const _Float16* gA0 = A  + (bm + srow) * K + sseg;
  const _Float16* gA1 = gA0 + (size_t)32 * K;
  const _Float16* gB0 = Bt + (bn + srow) * K + sseg;
  const _Float16* gB1 = gB0 + (size_t)32 * K;
  const int l0 = srow * 32 + sseg;
  const int l1 = (srow + 32) * 32 + sseg;

  auto stage = [&](int b) {
#if HAVE_ASYNC_LDS
    __builtin_amdgcn_global_load_async_to_lds_b128(
        (ASG v4i*)(void*)gA0, (ASL v4i*)(void*)&As[b][l0], 0, 0);
    __builtin_amdgcn_global_load_async_to_lds_b128(
        (ASG v4i*)(void*)gA1, (ASL v4i*)(void*)&As[b][l1], 0, 0);
    __builtin_amdgcn_global_load_async_to_lds_b128(
        (ASG v4i*)(void*)gB0, (ASL v4i*)(void*)&Bs[b][l0], 0, 0);
    __builtin_amdgcn_global_load_async_to_lds_b128(
        (ASG v4i*)(void*)gB1, (ASL v4i*)(void*)&Bs[b][l1], 0, 0);
#else
    *(v8h*)&As[b][l0] = *(const v8h*)gA0;
    *(v8h*)&As[b][l1] = *(const v8h*)gA1;
    *(v8h*)&Bs[b][l0] = *(const v8h*)gB0;
    *(v8h*)&Bs[b][l1] = *(const v8h*)gB1;
#endif
    __builtin_prefetch(gA0 + 32, 0, 1);   // speculative: dropped if invalid
    gA0 += 32; gA1 += 32; gB0 += 32; gB1 += 32;
  };

  stage(0);                                // prologue: tile 0 in flight
  int buf = 0;
  for (int k0 = 0; k0 < K; k0 += 32) {
    if (k0 + 32 < K) {
      stage(buf ^ 1);                      // overlap: tile k+1 -> other buffer
#if HAVE_ASYNC_LDS && HAVE_WAIT_ASYNC
      __builtin_amdgcn_s_wait_asynccnt(4); // drain previous batch only
#endif
    } else {
#if HAVE_ASYNC_LDS && HAVE_WAIT_ASYNC
      __builtin_amdgcn_s_wait_asynccnt(0);
#endif
    }
    __syncthreads();
    const _Float16* ar = &As[buf][(wave * 16 + nloc) * 32 + hi8];
    v16h a = load2x8(ar, ar + 16);
    #pragma unroll
    for (int nt = 0; nt < 4; ++nt) {
      const _Float16* br = &Bs[buf][(nt * 16 + nloc) * 32 + hi8];
      v16h b = load2x8(br, br + 16);
      acc[nt] = wmma_f16(a, b, acc[nt]);
    }
    __syncthreads();                       // protect buf before next stage
    buf ^= 1;
  }

  #pragma unroll
  for (int nt = 0; nt < 4; ++nt) {
    const int n = (int)bn + nt * 16 + nloc;
    const float bv = bias ? bias[n] : 0.f;
    #pragma unroll
    for (int r = 0; r < 8; ++r) {
      int m = (int)bm + wave * 16 + r + hi8;   // C layout: hi half rows +8
      float v = acc[nt][r] + bv;
      if (act == 1) {  // tanh-approx GELU
        float u = v;
        v = 0.5f * u * (1.f + tanhf(0.7978845608028654f *
                                    (u + 0.044715f * u * u * u)));
      }
      if (residual) v += residual[(size_t)m * N + n];
      if (C)   C[(size_t)m * N + n] = v;
      if (C16) C16[(size_t)m * N + n] = (_Float16)v;
    }
  }
}

// ---------------------------------------------------------------------------
// qkv (T,3,H,DH) f32 -> Q (H,T,DH) f16 (pre-scaled by DH^-0.5),
//                       K (H,T,DH) f16, Vt (H,DH,T) f16
// ---------------------------------------------------------------------------
__global__ void repack_qkv_kernel(const float* __restrict__ qkv,
                                  _Float16* __restrict__ Q,
                                  _Float16* __restrict__ K,
                                  _Float16* __restrict__ Vt) {
  int i = blockIdx.x * blockDim.x + threadIdx.x;
  if (i >= TSEQ * D_MODEL) return;
  int t = i / D_MODEL, hd = i % D_MODEL;
  int h = hd / DHEAD, d = hd % DHEAD;
  const float* base = qkv + (size_t)t * 3 * D_MODEL;
  size_t row = (size_t)h * TSEQ * DHEAD + (size_t)t * DHEAD + d;
  Q[row] = (_Float16)(base[hd] * 0.125f);
  K[row] = (_Float16)base[D_MODEL + hd];
  Vt[(size_t)h * DHEAD * TSEQ + (size_t)d * TSEQ + t] =
      (_Float16)base[2 * D_MODEL + hd];
}

// geo_v (T,D) f32 -> (H,DH,T) f16 transposed
__global__ void repack_gv_kernel(const float* __restrict__ gv,
                                 _Float16* __restrict__ Gt) {
  int i = blockIdx.x * blockDim.x + threadIdx.x;
  if (i >= TSEQ * D_MODEL) return;
  int t = i / D_MODEL, hd = i % D_MODEL;
  int h = hd / DHEAD, d = hd % DHEAD;
  Gt[(size_t)h * DHEAD * TSEQ + (size_t)d * TSEQ + t] = (_Float16)gv[i];
}

// ---------------------------------------------------------------------------
// Plucker lines: read_lines (T,H,6), J6-transformed write lines (T,H,6)
// ---------------------------------------------------------------------------
__global__ void lines_kernel(const float* __restrict__ ln,
                             const float* __restrict__ w1w,
                             const float* __restrict__ w2w,
                             const float* __restrict__ w1r,
                             const float* __restrict__ w2r,
                             float* __restrict__ rdl,
                             float* __restrict__ jwl) {
  int i = blockIdx.x * blockDim.x + threadIdx.x;
  if (i >= TSEQ * NHEAD) return;
  int t = i / NHEAD, h = i % NHEAD;
  const float* xt = ln + (size_t)t * D_MODEL;
  const float* xp = (t > 0) ? (ln + (size_t)(t - 1) * D_MODEL) : nullptr;
  float a1[4] = {}, a2[4] = {}, b1[4] = {}, b2[4] = {};
  const int c0 = h * 4;
  for (int k = 0; k < D_MODEL; ++k) {
    float xv = xt[k];
    float xpv = xp ? xp[k] : 0.f;
    size_t roff = (size_t)k * (4 * NHEAD) + c0;
    #pragma unroll
    for (int c = 0; c < 4; ++c) {
      a1[c] += xpv * w1w[roff + c];
      a2[c] += xv  * w2w[roff + c];
      b1[c] += xv  * w1r[roff + c];
      b2[c] += xv  * w2r[roff + c];
    }
  }
  float Lw[6] = { a1[0]*a2[1]-a1[1]*a2[0], a1[0]*a2[2]-a1[2]*a2[0],
                  a1[0]*a2[3]-a1[3]*a2[0], a1[1]*a2[2]-a1[2]*a2[1],
                  a1[1]*a2[3]-a1[3]*a2[1], a1[2]*a2[3]-a1[3]*a2[2] };
  float Lr[6] = { b1[0]*b2[1]-b1[1]*b2[0], b1[0]*b2[2]-b1[2]*b2[0],
                  b1[0]*b2[3]-b1[3]*b2[0], b1[1]*b2[2]-b1[2]*b2[1],
                  b1[1]*b2[3]-b1[3]*b2[1], b1[2]*b2[3]-b1[3]*b2[2] };
  float nw = 0.f, nr = 0.f;
  #pragma unroll
  for (int c = 0; c < 6; ++c) { nw += Lw[c]*Lw[c]; nr += Lr[c]*Lr[c]; }
  nw = 1.f / fmaxf(sqrtf(nw), 1e-12f);
  nr = 1.f / fmaxf(sqrtf(nr), 1e-12f);
  float* rp = rdl + (size_t)i * 6;
  float* jp = jwl + (size_t)i * 6;
  #pragma unroll
  for (int c = 0; c < 6; ++c) rp[c] = Lr[c] * nr;
  jp[0] =  Lw[5] * nw; jp[1] = -Lw[4] * nw; jp[2] =  Lw[3] * nw;
  jp[3] =  Lw[2] * nw; jp[4] = -Lw[1] * nw; jp[5] =  Lw[0] * nw;
}

// ---------------------------------------------------------------------------
// gate[t] = mean_h sigmoid(ln[t] @ gate_w + gate_b)
// ---------------------------------------------------------------------------
__global__ void gate_kernel(const float* __restrict__ ln,
                            const float* __restrict__ gw,
                            const float* __restrict__ gb,
                            float* __restrict__ gate) {
  int t = blockIdx.x * blockDim.x + threadIdx.x;
  if (t >= TSEQ) return;
  float acc[NHEAD] = {};
  const float* xt = ln + (size_t)t * D_MODEL;
  for (int k = 0; k < D_MODEL; ++k) {
    float xv = xt[k];
    #pragma unroll
    for (int h = 0; h < NHEAD; ++h) acc[h] += xv * gw[k * NHEAD + h];
  }
  float s = 0.f;
  #pragma unroll
  for (int h = 0; h < NHEAD; ++h)
    s += 1.f / (1.f + __expf(-(acc[h] + gb[h])));
  gate[t] = s / NHEAD;
}

// ---------------------------------------------------------------------------
// Dual-path flash attention; one wave per (16 query rows, head).
// ---------------------------------------------------------------------------
__global__ void attn_kernel(const _Float16* __restrict__ Qh,
                            const _Float16* __restrict__ Kh,
                            const _Float16* __restrict__ Vt,
                            const _Float16* __restrict__ GVt,
                            const float* __restrict__ rdl,
                            const float* __restrict__ jwl,
                            const float* __restrict__ inc_scale,
                            float* __restrict__ std_out,
                            float* __restrict__ geo_out) {
  __shared__ _Float16 pstd[16 * 32];
  __shared__ _Float16 pgeo[16 * 32];
  const int lane = threadIdx.x & 31;
  const int h = blockIdx.y;
  const int qbase = blockIdx.x * 16;
  const int nloc = lane & 15;
  const int hi8 = (lane >> 4) << 3;
  const _Float16* Q  = Qh  + (size_t)h * TSEQ * DHEAD;
  const _Float16* Kp = Kh  + (size_t)h * TSEQ * DHEAD;
  const _Float16* Vp = Vt  + (size_t)h * DHEAD * TSEQ;
  const _Float16* Gp = GVt + (size_t)h * DHEAD * TSEQ;

  const _Float16* qr = Q + (size_t)(qbase + nloc) * DHEAD;
  v16h qa0 = load2x8(qr + hi8, qr + hi8 + 16);
  v16h qa1 = load2x8(qr + 32 + hi8, qr + 32 + hi8 + 16);

  const float inc = inc_scale[h];
  float rl[8][6];
  #pragma unroll
  for (int r = 0; r < 8; ++r) {
    const float* rp = rdl + ((size_t)(qbase + r + hi8) * NHEAD + h) * 6;
    #pragma unroll
    for (int c = 0; c < 6; ++c) rl[r][c] = rp[c] * inc;
  }

  v8f zero8 = {};
  v8f o_s[4] = {zero8, zero8, zero8, zero8};
  v8f o_g[4] = {zero8, zero8, zero8, zero8};
  float m_s[8], l_s[8], m_g[8], l_g[8];
  #pragma unroll
  for (int r = 0; r < 8; ++r) {
    m_s[r] = -1e30f; l_s[r] = 0.f;
    m_g[r] = -1e30f; l_g[r] = 0.f;
  }

  for (int j0 = 0; j0 < qbase + 16; j0 += 32) {
    v8f s0 = zero8, s1 = zero8;
    {
      const _Float16* kr0 = Kp + (size_t)(j0 + nloc) * DHEAD;
      const _Float16* kr1 = Kp + (size_t)(j0 + 16 + nloc) * DHEAD;
      s0 = wmma_f16(qa0, load2x8(kr0 + hi8, kr0 + hi8 + 16), s0);
      s1 = wmma_f16(qa0, load2x8(kr1 + hi8, kr1 + hi8 + 16), s1);
      s0 = wmma_f16(qa1, load2x8(kr0 + 32 + hi8, kr0 + 32 + hi8 + 16), s0);
      s1 = wmma_f16(qa1, load2x8(kr1 + 32 + hi8, kr1 + 32 + hi8 + 16), s1);
    }
    float jw0[6], jw1[6];
    {
      const float* jp0 = jwl + ((size_t)(j0 + nloc) * NHEAD + h) * 6;
      const float* jp1 = jwl + ((size_t)(j0 + 16 + nloc) * NHEAD + h) * 6;
      #pragma unroll
      for (int c = 0; c < 6; ++c) { jw0[c] = jp0[c]; jw1[c] = jp1[c]; }
    }
    #pragma unroll
    for (int r = 0; r < 8; ++r) {
      const int row = qbase + r + hi8;
      float gg0 = 0.f, gg1 = 0.f;
      #pragma unroll
      for (int c = 0; c < 6; ++c) { gg0 += rl[r][c]*jw0[c]; gg1 += rl[r][c]*jw1[c]; }
      const bool ok0 = (j0 + nloc) <= row;
      const bool ok1 = (j0 + 16 + nloc) <= row;
      float ss0 = ok0 ? s0[r] : -1e30f;
      float ss1 = ok1 ? s1[r] : -1e30f;
      gg0 = ok0 ? gg0 : -1e30f;
      gg1 = ok1 ? gg1 : -1e30f;
      float mn = fmaxf(m_s[r], rowmax16(fmaxf(ss0, ss1)));
      float alpha = __expf(m_s[r] - mn);
      float p0 = __expf(ss0 - mn), p1 = __expf(ss1 - mn);
      l_s[r] = l_s[r] * alpha + rowsum16(p0 + p1);
      m_s[r] = mn;
      #pragma unroll
      for (int nt = 0; nt < 4; ++nt) o_s[nt][r] *= alpha;
      pstd[(r + hi8) * 32 + nloc]      = (_Float16)p0;
      pstd[(r + hi8) * 32 + 16 + nloc] = (_Float16)p1;
      float mng = fmaxf(m_g[r], rowmax16(fmaxf(gg0, gg1)));
      float ag = __expf(m_g[r] - mng);
      float q0 = __expf(gg0 - mng), q1 = __expf(gg1 - mng);
      l_g[r] = l_g[r] * ag + rowsum16(q0 + q1);
      m_g[r] = mng;
      #pragma unroll
      for (int nt = 0; nt < 4; ++nt) o_g[nt][r] *= ag;
      pgeo[(r + hi8) * 32 + nloc]      = (_Float16)q0;
      pgeo[(r + hi8) * 32 + 16 + nloc] = (_Float16)q1;
    }
    __syncthreads();
    v16h pa = load2x8(pstd + nloc * 32 + hi8, pstd + nloc * 32 + hi8 + 16);
    v16h ga = load2x8(pgeo + nloc * 32 + hi8, pgeo + nloc * 32 + hi8 + 16);
    #pragma unroll
    for (int nt = 0; nt < 4; ++nt) {
      const _Float16* vr = Vp + (size_t)(nt * 16 + nloc) * TSEQ + j0;
      const _Float16* gr = Gp + (size_t)(nt * 16 + nloc) * TSEQ + j0;
      o_s[nt] = wmma_f16(pa, load2x8(vr + hi8, vr + hi8 + 16), o_s[nt]);
      o_g[nt] = wmma_f16(ga, load2x8(gr + hi8, gr + hi8 + 16), o_g[nt]);
    }
    __syncthreads();
  }
  #pragma unroll
  for (int r = 0; r < 8; ++r) {
    const int row = qbase + r + hi8;
    const float inv_s = 1.f / l_s[r];
    const float inv_g = 1.f / l_g[r];
    #pragma unroll
    for (int nt = 0; nt < 4; ++nt) {
      int d = nt * 16 + nloc;
      std_out[(size_t)row * D_MODEL + h * DHEAD + d] = o_s[nt][r] * inv_s;
      geo_out[(size_t)row * D_MODEL + h * DHEAD + d] = o_g[nt][r] * inv_g;
    }
  }
}

// ---------------------------------------------------------------------------
__global__ void combine_kernel(const float* __restrict__ std_out,
                               const float* __restrict__ geo_out,
                               const float* __restrict__ gate,
                               _Float16* __restrict__ comb16) {
  int i = blockIdx.x * blockDim.x + threadIdx.x;
  if (i >= TSEQ * D_MODEL) return;
  float g = gate[i / D_MODEL];
  comb16[i] = (_Float16)((1.f - g) * std_out[i] + g * geo_out[i]);
}

// ---------------------------------------------------------------------------
extern "C" void kernel_launch(void* const* d_in, const int* in_sizes, int n_in,
                              void* d_out, int out_size, void* d_ws, size_t ws_size,
                              hipStream_t stream) {
  (void)in_sizes; (void)n_in; (void)out_size; (void)ws_size;
  const float* x      = (const float*)d_in[0];
  const float* ln1_g  = (const float*)d_in[1];
  const float* ln1_b  = (const float*)d_in[2];
  const float* qkv_w  = (const float*)d_in[3];
  const float* qkv_b  = (const float*)d_in[4];
  const float* w1w    = (const float*)d_in[5];
  const float* w2w    = (const float*)d_in[6];
  const float* w1r    = (const float*)d_in[7];
  const float* w2r    = (const float*)d_in[8];
  const float* gvw    = (const float*)d_in[9];
  const float* gvb    = (const float*)d_in[10];
  const float* gate_w = (const float*)d_in[11];
  const float* gate_b = (const float*)d_in[12];
  const float* inc    = (const float*)d_in[13];
  const float* out_w  = (const float*)d_in[14];
  const float* out_b  = (const float*)d_in[15];
  const float* ln2_g  = (const float*)d_in[16];
  const float* ln2_b  = (const float*)d_in[17];
  const float* fc_w   = (const float*)d_in[18];
  const float* fc_b   = (const float*)d_in[19];
  const float* proj_w = (const float*)d_in[20];
  const float* proj_b = (const float*)d_in[21];
  float* out = (float*)d_out;

  char* ws = (char*)d_ws;
  size_t off = 0;
  auto alloc = [&](size_t bytes) -> void* {
    void* p = ws + off;
    off = (off + bytes + 255) & ~(size_t)255;
    return p;
  };
  const size_t TD = (size_t)TSEQ * D_MODEL;
  float*    ln1f  = (float*)alloc(TD * 4);
  _Float16* ln1h  = (_Float16*)alloc(TD * 2);
  float*    qkvf  = (float*)alloc(TD * 3 * 4);
  _Float16* Qh    = (_Float16*)alloc(TD * 2);
  _Float16* Kh    = (_Float16*)alloc(TD * 2);
  _Float16* Vth   = (_Float16*)alloc(TD * 2);
  float*    gvf   = (float*)alloc(TD * 4);
  _Float16* GVth  = (_Float16*)alloc(TD * 2);
  float*    rdl   = (float*)alloc((size_t)TSEQ * NHEAD * 6 * 4);
  float*    jwl   = (float*)alloc((size_t)TSEQ * NHEAD * 6 * 4);
  float*    gatev = (float*)alloc((size_t)TSEQ * 4);
  float*    stdo  = (float*)alloc(TD * 4);
  float*    geoo  = (float*)alloc(TD * 4);
  _Float16* comb  = (_Float16*)alloc(TD * 2);
  float*    hbuf  = (float*)alloc(TD * 4);
  float*    mf    = (float*)alloc(TD * 4);
  _Float16* mh    = (_Float16*)alloc(TD * 2);
  _Float16* gelu16= (_Float16*)alloc((size_t)TSEQ * 4 * D_MODEL * 2);
  _Float16* qkvwT = (_Float16*)alloc((size_t)D_MODEL * 3 * D_MODEL * 2);
  _Float16* gvwT  = (_Float16*)alloc((size_t)D_MODEL * D_MODEL * 2);
  _Float16* outwT = (_Float16*)alloc((size_t)D_MODEL * D_MODEL * 2);
  _Float16* fcwT  = (_Float16*)alloc((size_t)D_MODEL * 4 * D_MODEL * 2);
  _Float16* pwT   = (_Float16*)alloc((size_t)4 * D_MODEL * D_MODEL * 2);

  auto cvtT = [&](const float* src, _Float16* dst, int K, int N) {
    transpose_w_kernel<<<(K * N + 255) / 256, 256, 0, stream>>>(src, dst, K, N);
  };
  cvtT(qkv_w,  qkvwT, D_MODEL, 3 * D_MODEL);
  cvtT(gvw,    gvwT,  D_MODEL, D_MODEL);
  cvtT(out_w,  outwT, D_MODEL, D_MODEL);
  cvtT(fc_w,   fcwT,  D_MODEL, 4 * D_MODEL);
  cvtT(proj_w, pwT,   4 * D_MODEL, D_MODEL);

  layernorm_kernel<<<TSEQ, 256, 0, stream>>>(x, ln1_g, ln1_b, ln1f, ln1h);

  gemm_kernel<<<dim3(TSEQ / GBM, (3 * D_MODEL) / GBN), 128, 0, stream>>>(
      ln1h, qkvwT, qkv_b, nullptr, qkvf, nullptr, TSEQ, 3 * D_MODEL, D_MODEL, 0);
  repack_qkv_kernel<<<(TSEQ * D_MODEL + 255) / 256, 256, 0, stream>>>(qkvf, Qh, Kh, Vth);

  gemm_kernel<<<dim3(TSEQ / GBM, D_MODEL / GBN), 128, 0, stream>>>(
      ln1h, gvwT, gvb, nullptr, gvf, nullptr, TSEQ, D_MODEL, D_MODEL, 0);
  repack_gv_kernel<<<(TSEQ * D_MODEL + 255) / 256, 256, 0, stream>>>(gvf, GVth);

  lines_kernel<<<(TSEQ * NHEAD + 255) / 256, 256, 0, stream>>>(
      ln1f, w1w, w2w, w1r, w2r, rdl, jwl);
  gate_kernel<<<(TSEQ + 255) / 256, 256, 0, stream>>>(ln1f, gate_w, gate_b, gatev);

  attn_kernel<<<dim3(TSEQ / 16, NHEAD), 32, 0, stream>>>(
      Qh, Kh, Vth, GVth, rdl, jwl, inc, stdo, geoo);

  combine_kernel<<<(TSEQ * D_MODEL + 255) / 256, 256, 0, stream>>>(stdo, geoo, gatev, comb);

  gemm_kernel<<<dim3(TSEQ / GBM, D_MODEL / GBN), 128, 0, stream>>>(
      comb, outwT, out_b, x, hbuf, nullptr, TSEQ, D_MODEL, D_MODEL, 0);

  layernorm_kernel<<<TSEQ, 256, 0, stream>>>(hbuf, ln2_g, ln2_b, mf, mh);

  gemm_kernel<<<dim3(TSEQ / GBM, (4 * D_MODEL) / GBN), 128, 0, stream>>>(
      mh, fcwT, fc_b, nullptr, nullptr, gelu16, TSEQ, 4 * D_MODEL, D_MODEL, 1);

  gemm_kernel<<<dim3(TSEQ / GBM, D_MODEL / GBN), 128, 0, stream>>>(
      gelu16, pwT, proj_b, hbuf, out, nullptr, TSEQ, D_MODEL, 4 * D_MODEL, 0);
}
